// selfattention_43078521979570
// MI455X (gfx1250) — matrix operand order repaired and checked by reading, not compiled, not run on hardware
//
#include <hip/hip_runtime.h>

// MI455X (gfx1250) self-attention, WMMA f16->f32 path.
// Pipeline: 3x QKV projection GEMMs -> causal scores GEMM -> row softmax -> P@V GEMM.
// Workspace: Qh/Kh/Vh f16 (3 x 16MB) + scores/probs f16 (32MB) = 80MB.
// Round 4: replace __syncthreads() in the GEMM loops with an LDS-only split
// barrier (s_wait_dscnt 0 ; s_barrier_signal/wait -1) so the compiler does not
// drain loadcnt at the barrier — global loads stay in flight across the WMMA
// block and are only waited at the commit (cvt+ds_store) site.

typedef __attribute__((ext_vector_type(16))) _Float16 v16h;
typedef __attribute__((ext_vector_type(8)))  _Float16 v8h;
typedef __attribute__((ext_vector_type(8)))  float    v8f;

#define S_LEN  2048
#define D_DIM  1024
#define NBATCH 4
#define LDA    40   // LDS row stride in halves: 80B, keeps 16B alignment for b128 loads

__device__ __forceinline__ v8f zero8() {
  v8f z;
#pragma unroll
  for (int r = 0; r < 8; ++r) z[r] = 0.0f;
  return z;
}

// Workgroup barrier that only drains LDS (DScnt), not global loads (LOADcnt).
// Safe here: the only cross-wave state at these barriers is LDS; global loads
// land in private VGPRs. Async-LDS writes are drained by wait_async0() first.
__device__ __forceinline__ void lds_barrier() {
  asm volatile("s_wait_dscnt 0x0\n\t"
               "s_barrier_signal -1\n\t"
               "s_barrier_wait -1" ::: "memory");
}

// Async global->LDS 16B copy, tracked by ASYNCcnt (CDNA5 ISA).
// Flat shared-pointer low 32 bits == LDS byte offset (aperture truncation rule).
__device__ __forceinline__ void async_b128(const _Float16* g, const _Float16* l) {
  unsigned loff = (unsigned)(size_t)l;
  asm volatile("global_load_async_to_lds_b128 %0, %1, off"
               :: "v"(loff), "v"((unsigned long long)(size_t)g)
               : "memory");
}
__device__ __forceinline__ void wait_async0() {
  asm volatile("s_wait_asynccnt 0x0" ::: "memory");
}

// A fragment 16x32 f16 from LDS tile (row-major, stride LDA halves).
// lane 0-15: M=lane, K=0..7 (v0-3) & 16..23 (v4-7); lane 16-31: M=lane-16, K=8..15 & 24..31.
__device__ __forceinline__ v16h frag_a16(const _Float16* base, int lane) {
  const int hl = lane & 15, hi = lane >> 4;
  const _Float16* p = base + hl * LDA + hi * 8;
  v8h lo = *(const v8h*)p;
  v8h hh = *(const v8h*)(p + 16);
  return __builtin_shufflevector(lo, hh, 0,1,2,3,4,5,6,7,8,9,10,11,12,13,14,15);
}

// B fragment 32x16 f16 from LDS stored as [n][k] (stride LDA halves).
// lane 0-15: N=lane, K=0..7 & 16..23; lane 16-31: N=lane-16, K=8..15 & 24..31.
__device__ __forceinline__ v16h frag_b16(const _Float16* base, int lane) {
  const int nl = lane & 15, hi = lane >> 4;
  const _Float16* p = base + nl * LDA + hi * 8;
  v8h lo = *(const v8h*)p;
  v8h hh = *(const v8h*)(p + 16);
  return __builtin_shufflevector(lo, hh, 0,1,2,3,4,5,6,7,8,9,10,11,12,13,14,15);
}

__device__ __forceinline__ v8f wmma16(v16h a, v16h b, v8f c) {
  return __builtin_amdgcn_wmma_f32_16x16x32_f16(false, a, false, b, (short)0, c, false, false);
}

// ---------------- Kernel 1: Out[m][n] = f16( X[m][:] . W[n][:] + bias[n] ) -----------
// M=8192, N=1024, K=1024. Block tile 128x128, 8 waves, wave tile 32x64 (2x4 WMMA).
// f32->f16 conversion in flight => register-staged, double-buffered LDS,
// software-pipelined: issue loads(t+2) / compute(t) / commit(t+1).
__global__ __launch_bounds__(256) void proj_kernel(
    const float* __restrict__ X, const float* __restrict__ W,
    const float* __restrict__ bias, _Float16* __restrict__ Out) {
  __shared__ __align__(16) _Float16 As[2][128 * LDA];
  __shared__ __align__(16) _Float16 Bs[2][128 * LDA];
  const int tid = threadIdx.x;
  const int lane = tid & 31, wv = tid >> 5;
  const int wm = wv & 3, wn = wv >> 2;
  const int m0 = blockIdx.y * 128;
  const int n0 = blockIdx.x * 128;
  const int srow = tid >> 1;
  const int skk  = (tid & 1) * 16;

  float4 xa[4], wb[4];
  auto gload = [&](int k0) {
    const float4* xp = (const float4*)(X + (size_t)(m0 + srow) * D_DIM + k0 + skk);
    const float4* wp = (const float4*)(W + (size_t)(n0 + srow) * D_DIM + k0 + skk);
#pragma unroll
    for (int q = 0; q < 4; ++q) { xa[q] = xp[q]; wb[q] = wp[q]; }
  };
  auto commit = [&](int buf) {
    v8h ah0 = {(_Float16)xa[0].x,(_Float16)xa[0].y,(_Float16)xa[0].z,(_Float16)xa[0].w,
               (_Float16)xa[1].x,(_Float16)xa[1].y,(_Float16)xa[1].z,(_Float16)xa[1].w};
    v8h ah1 = {(_Float16)xa[2].x,(_Float16)xa[2].y,(_Float16)xa[2].z,(_Float16)xa[2].w,
               (_Float16)xa[3].x,(_Float16)xa[3].y,(_Float16)xa[3].z,(_Float16)xa[3].w};
    v8h bh0 = {(_Float16)wb[0].x,(_Float16)wb[0].y,(_Float16)wb[0].z,(_Float16)wb[0].w,
               (_Float16)wb[1].x,(_Float16)wb[1].y,(_Float16)wb[1].z,(_Float16)wb[1].w};
    v8h bh1 = {(_Float16)wb[2].x,(_Float16)wb[2].y,(_Float16)wb[2].z,(_Float16)wb[2].w,
               (_Float16)wb[3].x,(_Float16)wb[3].y,(_Float16)wb[3].z,(_Float16)wb[3].w};
    *(v8h*)&As[buf][srow * LDA + skk]     = ah0;
    *(v8h*)&As[buf][srow * LDA + skk + 8] = ah1;
    *(v8h*)&Bs[buf][srow * LDA + skk]     = bh0;
    *(v8h*)&Bs[buf][srow * LDA + skk + 8] = bh1;
  };

  v8f acc[2][4];
#pragma unroll
  for (int i = 0; i < 2; ++i)
#pragma unroll
    for (int j = 0; j < 4; ++j) acc[i][j] = zero8();

  gload(0);
  commit(0);
  gload(32);
  int cur = 0;
  for (int k0 = 0; k0 < D_DIM; k0 += 32, cur ^= 1) {
    lds_barrier();                           // buf `cur` published; `cur^1` free
    v16h afr[2];
#pragma unroll
    for (int tm = 0; tm < 2; ++tm) afr[tm] = frag_a16(&As[cur][(32 * wm + 16 * tm) * LDA], lane);
#pragma unroll
    for (int tn = 0; tn < 4; ++tn) {
      v16h bfr = frag_b16(&Bs[cur][(64 * wn + 16 * tn) * LDA], lane);
      acc[0][tn] = wmma16(afr[0], bfr, acc[0][tn]);
      acc[1][tn] = wmma16(afr[1], bfr, acc[1][tn]);
    }
    if (k0 + 32 < D_DIM) {
      commit(cur ^ 1);                       // loadcnt wait lands after the WMMAs
      if (k0 + 64 < D_DIM) gload(k0 + 64);   // issue loads for tile t+2
    }
  }

  const int hl = lane & 15, hi = lane >> 4;
#pragma unroll
  for (int tn = 0; tn < 4; ++tn) {
    const int nn = n0 + 64 * wn + 16 * tn + hl;
    const float bv = bias[nn];
#pragma unroll
    for (int tm = 0; tm < 2; ++tm) {
      const int mb = m0 + 32 * wm + 16 * tm + 8 * hi;
#pragma unroll
      for (int r = 0; r < 8; ++r)
        Out[(size_t)(mb + r) * D_DIM + nn] = (_Float16)(acc[tm][tn][r] + bv);
    }
  }
}

// ---------------- Kernel 2: S[i][j] = scale * Q[i].K[j], causal mask, f16 ------------
// Pure f16 copies: async global->LDS staging, double-buffered.
__global__ __launch_bounds__(256) void scores_kernel(
    const _Float16* __restrict__ Qh, const _Float16* __restrict__ Kh,
    _Float16* __restrict__ Sc) {
  const int jt = blockIdx.x, it = blockIdx.y, b = blockIdx.z;
  if (jt > it) return;  // uniform per block: WMMA waves keep EXEC all-ones
  __shared__ __align__(16) _Float16 As[2][128 * LDA];
  __shared__ __align__(16) _Float16 Bs[2][128 * LDA];
  const int tid = threadIdx.x;
  const int lane = tid & 31, wv = tid >> 5;
  const int wm = wv & 3, wn = wv >> 2;
  const int i0 = it * 128, j0 = jt * 128;
  const int srow = tid >> 1;
  const int skk  = (tid & 1) * 16;
  const _Float16* Q = Qh + (size_t)b * S_LEN * D_DIM;
  const _Float16* K = Kh + (size_t)b * S_LEN * D_DIM;

  auto stage = [&](int k0, int buf) {
    const _Float16* q = Q + (size_t)(i0 + srow) * D_DIM + k0 + skk;
    const _Float16* k = K + (size_t)(j0 + srow) * D_DIM + k0 + skk;
    async_b128(q,     &As[buf][srow * LDA + skk]);
    async_b128(q + 8, &As[buf][srow * LDA + skk + 8]);
    async_b128(k,     &Bs[buf][srow * LDA + skk]);
    async_b128(k + 8, &Bs[buf][srow * LDA + skk + 8]);
  };

  v8f acc[2][4];
#pragma unroll
  for (int i = 0; i < 2; ++i)
#pragma unroll
    for (int j = 0; j < 4; ++j) acc[i][j] = zero8();

  stage(0, 0);
  int cur = 0;
  for (int k0 = 0; k0 < D_DIM; k0 += 32, cur ^= 1) {
    wait_async0();                           // my async writes into buf `cur` done
    lds_barrier();                           // everyone's staging of `cur` visible
    if (k0 + 32 < D_DIM) stage(k0 + 32, cur ^ 1);
    v16h afr[2];
#pragma unroll
    for (int tm = 0; tm < 2; ++tm) afr[tm] = frag_a16(&As[cur][(32 * wm + 16 * tm) * LDA], lane);
#pragma unroll
    for (int tn = 0; tn < 4; ++tn) {
      v16h bfr = frag_b16(&Bs[cur][(64 * wn + 16 * tn) * LDA], lane);
      acc[0][tn] = wmma16(afr[0], bfr, acc[0][tn]);
      acc[1][tn] = wmma16(afr[1], bfr, acc[1][tn]);
    }
  }

  _Float16* S = Sc + (size_t)b * S_LEN * S_LEN;
  const int hl = lane & 15, hi = lane >> 4;
  const _Float16 neginf = (_Float16)(-__builtin_inff());
#pragma unroll
  for (int tn = 0; tn < 4; ++tn) {
    const int j = j0 + 64 * wn + 16 * tn + hl;
#pragma unroll
    for (int tm = 0; tm < 2; ++tm) {
      const int ib = i0 + 32 * wm + 16 * tm + 8 * hi;
#pragma unroll
      for (int r = 0; r < 8; ++r) {
        const int i = ib + r;
        const float v = acc[tm][tn][r] * 0.03125f;  // 1/sqrt(1024)
        S[(size_t)i * S_LEN + j] = (j <= i) ? (_Float16)v : neginf;
      }
    }
  }
}

// ---------------- Kernel 3: in-place causal row softmax (f16, fp32 math) -------------
__global__ __launch_bounds__(256) void softmax_kernel(_Float16* __restrict__ Sc) {
  __shared__ float red[256];
  const int tid = threadIdx.x;
  const int row = blockIdx.x;
  const int b = row >> 11, i = row & (S_LEN - 1);
  _Float16* S = Sc + (size_t)b * S_LEN * S_LEN + (size_t)i * S_LEN;
  const int jend = ((i >> 7) + 1) << 7;  // full 128-wide diag tile (masked = -inf -> 0)

  float m = -3.4e38f;
  for (int j = tid; j < jend; j += 256) m = fmaxf(m, (float)S[j]);
  red[tid] = m; __syncthreads();
  for (int s = 128; s > 0; s >>= 1) { if (tid < s) red[tid] = fmaxf(red[tid], red[tid + s]); __syncthreads(); }
  m = red[0]; __syncthreads();

  float sum = 0.0f;
  for (int j = tid; j < jend; j += 256) sum += __expf((float)S[j] - m);
  red[tid] = sum; __syncthreads();
  for (int s = 128; s > 0; s >>= 1) { if (tid < s) red[tid] += red[tid + s]; __syncthreads(); }
  sum = red[0];

  const float inv = 1.0f / sum;
  for (int j = tid; j < jend; j += 256) S[j] = (_Float16)(__expf((float)S[j] - m) * inv);
}

// ---------------- Kernel 4: O[i][d] = sum_j P[i][j] * V[j][d], fp32 out --------------
// P staged async; V staged via regs with split issue/commit (transpose to [d][j]).
__global__ __launch_bounds__(256) void attnv_kernel(
    const _Float16* __restrict__ Sc, const _Float16* __restrict__ Vh,
    float* __restrict__ Out) {
  __shared__ __align__(16) _Float16 Ps[2][128 * LDA];
  __shared__ __align__(16) _Float16 Vs[2][128 * LDA];  // stored [d][j]
  const int dt = blockIdx.x, it = blockIdx.y, b = blockIdx.z;
  const int tid = threadIdx.x;
  const int lane = tid & 31, wv = tid >> 5;
  const int wm = wv & 3, wn = wv >> 2;
  const int i0 = it * 128, d0 = dt * 128;
  const int srow = tid >> 1, skk = (tid & 1) * 16;
  const int vjj = tid >> 3, vdd = (tid & 7) * 16;
  const _Float16* P = Sc + (size_t)b * S_LEN * S_LEN;
  const _Float16* V = Vh + (size_t)b * S_LEN * D_DIM;

  v8h vr0, vr1;
  auto vload = [&](int j0v) {
    const _Float16* vp = V + (size_t)(j0v + vjj) * D_DIM + d0 + vdd;
    vr0 = *(const v8h*)vp;
    vr1 = *(const v8h*)(vp + 8);
  };
  auto vcommit = [&](int buf) {
#pragma unroll
    for (int q = 0; q < 8; ++q) {
      Vs[buf][(vdd + q) * LDA + vjj]     = vr0[q];
      Vs[buf][(vdd + 8 + q) * LDA + vjj] = vr1[q];
    }
  };
  auto pstage = [&](int j0v, int buf) {
    const _Float16* pp = P + (size_t)(i0 + srow) * S_LEN + j0v + skk;
    async_b128(pp,     &Ps[buf][srow * LDA + skk]);
    async_b128(pp + 8, &Ps[buf][srow * LDA + skk + 8]);
  };

  v8f acc[2][4];
#pragma unroll
  for (int i = 0; i < 2; ++i)
#pragma unroll
    for (int j = 0; j < 4; ++j) acc[i][j] = zero8();

  const int kend = i0 + 128;  // causal: only j-tiles <= i-tile (kend >= 128)
  vload(0);
  vcommit(0);
  pstage(0, 0);
  vload(32);
  int cur = 0;
  for (int j0 = 0; j0 < kend; j0 += 32, cur ^= 1) {
    wait_async0();
    lds_barrier();
    if (j0 + 32 < kend) pstage(j0 + 32, cur ^ 1);  // async issue before compute
    v16h afr[2];
#pragma unroll
    for (int tm = 0; tm < 2; ++tm) afr[tm] = frag_a16(&Ps[cur][(32 * wm + 16 * tm) * LDA], lane);
#pragma unroll
    for (int tn = 0; tn < 4; ++tn) {
      v16h bfr = frag_b16(&Vs[cur][(64 * wn + 16 * tn) * LDA], lane);
      acc[0][tn] = wmma16(afr[0], bfr, acc[0][tn]);
      acc[1][tn] = wmma16(afr[1], bfr, acc[1][tn]);
    }
    if (j0 + 32 < kend) {
      vcommit(cur ^ 1);                            // loadcnt wait after WMMAs
      if (j0 + 64 < kend) vload(j0 + 64);
    }
  }

  const int hl = lane & 15, hi = lane >> 4;
#pragma unroll
  for (int tn = 0; tn < 4; ++tn) {
    const int d = d0 + 64 * wn + 16 * tn + hl;
#pragma unroll
    for (int tm = 0; tm < 2; ++tm) {
      const int ib = i0 + 32 * wm + 16 * tm + 8 * hi;
#pragma unroll
      for (int r = 0; r < 8; ++r)
        Out[(size_t)(b * S_LEN + ib + r) * D_DIM + d] = acc[tm][tn][r];
    }
  }
}

extern "C" void kernel_launch(void* const* d_in, const int* in_sizes, int n_in,
                              void* d_out, int out_size, void* d_ws, size_t ws_size,
                              hipStream_t stream) {
  (void)in_sizes; (void)n_in; (void)out_size; (void)ws_size;
  const float* X  = (const float*)d_in[0];
  const float* Wq = (const float*)d_in[1];
  const float* bq = (const float*)d_in[2];
  const float* Wk = (const float*)d_in[3];
  const float* bk = (const float*)d_in[4];
  const float* Wv = (const float*)d_in[5];
  const float* bv = (const float*)d_in[6];
  float* Out = (float*)d_out;

  const size_t nqkv = (size_t)NBATCH * S_LEN * D_DIM;  // 8M f16 elems = 16MB each
  _Float16* Qh = (_Float16*)d_ws;
  _Float16* Kh = Qh + nqkv;
  _Float16* Vh = Kh + nqkv;
  _Float16* Sc = Vh + nqkv;  // 4 x 2048 x 2048 f16 = 32MB

  dim3 blk(256, 1, 1);
  proj_kernel<<<dim3(8, 64, 1), blk, 0, stream>>>(X, Wq, bq, Qh);
  proj_kernel<<<dim3(8, 64, 1), blk, 0, stream>>>(X, Wk, bk, Kh);
  proj_kernel<<<dim3(8, 64, 1), blk, 0, stream>>>(X, Wv, bv, Vh);
  scores_kernel<<<dim3(16, 16, 4), blk, 0, stream>>>(Qh, Kh, Sc);
  softmax_kernel<<<dim3(NBATCH * S_LEN, 1, 1), blk, 0, stream>>>(Sc);
  attnv_kernel<<<dim3(8, 16, 4), blk, 0, stream>>>(Sc, Vh, Out);
}
